// linear_attention_1322849927773
// MI455X (gfx1250) — matrix operand order repaired and checked by reading, not compiled, not run on hardware
//
#include <hip/hip_runtime.h>
#include <hip/hip_bf16.h>

// ---------------------------------------------------------------------------
// Types for WMMA (gfx1250, wave32): v_wmma_f32_16x16x32_f16
// ---------------------------------------------------------------------------
typedef _Float16 half_t;
typedef __attribute__((ext_vector_type(16))) _Float16 v16h;
typedef __attribute__((ext_vector_type(8)))  _Float16 v8h;
typedef __attribute__((ext_vector_type(8)))  float    v8f;

union Frag16 { v16h v; v8h h[2]; };

__device__ __forceinline__ v8f wmma_f16(const Frag16& a, const Frag16& b, v8f c) {
    // 8 args: (neg_a, A, neg_b, B, c_mod, C, reuse_a, reuse_b)
    return __builtin_amdgcn_wmma_f32_16x16x32_f16(
        false, a.v, false, b.v, (short)0, c, false, false);
}

// Problem constants (match reference)
#define BATCH   8
#define TDIM    8192L
#define DIMC    512
#define HEADS   8
#define DHEAD   64
#define PROWS   1536            // [qp(512); k(512); v(512)]
#define SCALE   0.125f          // 64^-0.5

// ---------------------------------------------------------------------------
// K0: f32 -> f16 conversion (8 elements / thread)
// ---------------------------------------------------------------------------
__global__ void cvt_f32_f16_kernel(const float* __restrict__ src,
                                   half_t* __restrict__ dst, long n) {
    long i = ((long)blockIdx.x * blockDim.x + threadIdx.x) * 8;
    if (i + 8 <= n) {
        float4 a = *(const float4*)(src + i);
        float4 b = *(const float4*)(src + i + 4);
        v8h o;
        o[0] = (half_t)a.x; o[1] = (half_t)a.y; o[2] = (half_t)a.z; o[3] = (half_t)a.w;
        o[4] = (half_t)b.x; o[5] = (half_t)b.y; o[6] = (half_t)b.z; o[7] = (half_t)b.w;
        *(v8h*)(dst + i) = o;
    }
}

// ---------------------------------------------------------------------------
// Generic batched GEMM:  C[b] = A[b](MxK,f16) * B[b](KxN,f16)  (+ bias[m])
// Block = 256 threads = 8 waves; tile 128(M) x 128(N) x 32(K).
// Wave (wm=wave&1, wn=wave>>1) owns a 64x32 sub-tile = 4x2 WMMA tiles.
// A staged row-major in LDS; B transposed on store so both operand frags are
// contiguous 16B LDS reads matching the ISA VGPR layouts.
// ---------------------------------------------------------------------------
template <typename OutT, bool BIAS>
__global__ void __launch_bounds__(256)
gemm_f16_kernel(const half_t* __restrict__ A, const half_t* __restrict__ B,
                OutT* __restrict__ C, const float* __restrict__ bias,
                int M, int N, int K, long sA, long sB, long sC) {
    __shared__ half_t As[128 * 32];   // row-major [m][k]
    __shared__ half_t Bt[128 * 32];   // transposed [n][k]

    const half_t* Ab = A + (long)blockIdx.z * sA;
    const half_t* Bb = B + (long)blockIdx.z * sB;
    OutT*         Cb = C + (long)blockIdx.z * sC;

    const int m0 = blockIdx.y * 128;
    const int n0 = blockIdx.x * 128;
    const int tid  = threadIdx.x;
    const int wave = tid >> 5;
    const int lane = tid & 31;
    const int laneM = lane & 15;
    const int sel   = lane >> 4;       // 0: lanes 0-15, 1: lanes 16-31
    const int wm = wave & 1;           // 2 waves along M
    const int wn = wave >> 1;          // 4 waves along N

    v8f acc[4][2];
    #pragma unroll
    for (int i = 0; i < 4; ++i)
        #pragma unroll
        for (int j = 0; j < 2; ++j) acc[i][j] = (v8f){0,0,0,0,0,0,0,0};

    for (int k0 = 0; k0 < K; k0 += 32) {
        // --- stage A tile (128x32): 512 v8h chunks, 2 per thread ---
        #pragma unroll
        for (int i = 0; i < 2; ++i) {
            int c   = tid + i * 256;
            int row = c >> 2;            // 4 chunks per row
            int col = (c & 3) << 3;
            const half_t* gp = Ab + (long)(m0 + row) * K + k0 + col;
            *(v8h*)(&As[row * 32 + col]) = *(const v8h*)gp;
            if (k0 + 32 < K) __builtin_prefetch(gp + 32, 0, 0);  // global_prefetch_b8
        }
        // --- stage B tile (32x128) transposed into Bt[n][k] ---
        #pragma unroll
        for (int i = 0; i < 2; ++i) {
            int c  = tid + i * 256;
            int kr = c >> 4;             // 16 chunks per B row
            int nc = (c & 15) << 3;
            const half_t* gp = Bb + (long)(k0 + kr) * N + n0 + nc;
            v8h b = *(const v8h*)gp;
            if (k0 + 32 < K) __builtin_prefetch(gp + 32L * N, 0, 0);
            #pragma unroll
            for (int j = 0; j < 8; ++j) Bt[(nc + j) * 32 + kr] = b[j];
        }
        __syncthreads();

        // --- fragments + 8 WMMA ---
        Frag16 af[4], bf[2];
        #pragma unroll
        for (int mt = 0; mt < 4; ++mt) {
            int row = wm * 64 + mt * 16 + laneM;
            af[mt].h[0] = *(const v8h*)(&As[row * 32 + sel * 8]);        // K 0..7 / 8..15
            af[mt].h[1] = *(const v8h*)(&As[row * 32 + sel * 8 + 16]);   // K 16..23 / 24..31
        }
        #pragma unroll
        for (int nt = 0; nt < 2; ++nt) {
            int row = wn * 32 + nt * 16 + laneM;
            bf[nt].h[0] = *(const v8h*)(&Bt[row * 32 + sel * 8]);        // K 0..7 / 8..15
            bf[nt].h[1] = *(const v8h*)(&Bt[row * 32 + sel * 8 + 16]);   // K 16..23 / 24..31
        }
        #pragma unroll
        for (int mt = 0; mt < 4; ++mt)
            #pragma unroll
            for (int nt = 0; nt < 2; ++nt)
                acc[mt][nt] = wmma_f16(af[mt], bf[nt], acc[mt][nt]);
        __syncthreads();
    }

    // --- epilogue: C/D layout = vgpr r -> row r + 8*sel, lane -> col ---
    #pragma unroll
    for (int mt = 0; mt < 4; ++mt) {
        #pragma unroll
        for (int nt = 0; nt < 2; ++nt) {
            #pragma unroll
            for (int r = 0; r < 8; ++r) {
                int m = m0 + wm * 64 + mt * 16 + sel * 8 + r;
                int n = n0 + wn * 32 + nt * 16 + laneM;
                float v = acc[mt][nt][r];
                if (BIAS) v += bias[m];
                Cb[(long)m * N + n] = (OutT)v;
            }
        }
    }
}

// ---------------------------------------------------------------------------
// K2: softmax over channel dim d (64), in place on P rows [0,512), times SCALE
// one thread per (b,h,t); loads are coalesced along t
// ---------------------------------------------------------------------------
__global__ void qh_softmax_kernel(half_t* __restrict__ P) {
    long idx = (long)blockIdx.x * blockDim.x + threadIdx.x;   // < B*H*T
    long t = idx % TDIM;
    long bh = idx / TDIM;
    int h = (int)(bh & 7);
    int b = (int)(bh >> 3);
    half_t* base = P + ((long)b * PROWS + h * DHEAD) * TDIM + t;
    float v[64];
    float m = -__builtin_inff();
    #pragma unroll
    for (int d = 0; d < 64; ++d) { v[d] = (float)base[(long)d * TDIM]; m = fmaxf(m, v[d]); }
    float s = 0.f;
    #pragma unroll
    for (int d = 0; d < 64; ++d) { v[d] = __expf(v[d] - m); s += v[d]; }
    float inv = SCALE / s;
    #pragma unroll
    for (int d = 0; d < 64; ++d) base[(long)d * TDIM] = (half_t)(v[d] * inv);
}

// ---------------------------------------------------------------------------
// K3: softmax over time (T=8192), in place on P rows [512,1024); 1 block/row
// ---------------------------------------------------------------------------
__global__ void kh_softmax_kernel(half_t* __restrict__ P) {
    int b   = blockIdx.x >> 9;
    int row = blockIdx.x & 511;
    half_t* p = P + ((long)b * PROWS + 512 + row) * TDIM;
    int tid = threadIdx.x;
    __shared__ float red[256];

    float vals[32];
    float m = -__builtin_inff();
    #pragma unroll
    for (int j = 0; j < 32; ++j) { vals[j] = (float)p[tid + j * 256]; m = fmaxf(m, vals[j]); }
    red[tid] = m; __syncthreads();
    for (int s = 128; s > 0; s >>= 1) {
        if (tid < s) red[tid] = fmaxf(red[tid], red[tid + s]);
        __syncthreads();
    }
    m = red[0]; __syncthreads();

    float sum = 0.f;
    #pragma unroll
    for (int j = 0; j < 32; ++j) { vals[j] = __expf(vals[j] - m); sum += vals[j]; }
    red[tid] = sum; __syncthreads();
    for (int s = 128; s > 0; s >>= 1) {
        if (tid < s) red[tid] += red[tid + s];
        __syncthreads();
    }
    float inv = 1.f / red[0];
    #pragma unroll
    for (int j = 0; j < 32; ++j) p[tid + j * 256] = (half_t)(vals[j] * inv);
}

// ---------------------------------------------------------------------------
// K5: context[b,h] (64x64 f32) = kh(64xT) * vh(64xT)^T  — WMMA, K=8192
// Both operands are K-contiguous in memory -> direct global frag loads.
// 1 block / (b,h); 8 waves: wave owns (mt = w&3, nt pair = (w>>2)*2)
// ---------------------------------------------------------------------------
__global__ void __launch_bounds__(256)
context_kernel(const half_t* __restrict__ P, float* __restrict__ ctx) {
    int bh = blockIdx.x;
    int b = bh >> 3, h = bh & 7;
    const half_t* kh = P + ((long)b * PROWS + 512  + h * DHEAD) * TDIM;
    const half_t* vh = P + ((long)b * PROWS + 1024 + h * DHEAD) * TDIM;
    int tid = threadIdx.x, wave = tid >> 5, lane = tid & 31;
    int laneM = lane & 15, sel = lane >> 4;
    int mt  = wave & 3;
    int ntp = (wave >> 2) * 2;

    const half_t* arow  = kh + (long)(mt * 16 + laneM) * TDIM;
    const half_t* brow0 = vh + (long)(ntp * 16 + laneM) * TDIM;
    const half_t* brow1 = vh + (long)((ntp + 1) * 16 + laneM) * TDIM;

    v8f acc0 = (v8f){0,0,0,0,0,0,0,0};
    v8f acc1 = (v8f){0,0,0,0,0,0,0,0};
    for (long k0 = 0; k0 < TDIM; k0 += 32) {
        Frag16 a, b0, b1;
        a.h[0]  = *(const v8h*)(arow  + k0 + sel * 8);
        a.h[1]  = *(const v8h*)(arow  + k0 + sel * 8 + 16);
        b0.h[0] = *(const v8h*)(brow0 + k0 + sel * 8);
        b0.h[1] = *(const v8h*)(brow0 + k0 + sel * 8 + 16);
        b1.h[0] = *(const v8h*)(brow1 + k0 + sel * 8);
        b1.h[1] = *(const v8h*)(brow1 + k0 + sel * 8 + 16);
        acc0 = wmma_f16(a, b0, acc0);
        acc1 = wmma_f16(a, b1, acc1);
    }
    float* out = ctx + (long)bh * 64 * 64;   // [d][e]
    #pragma unroll
    for (int r = 0; r < 8; ++r) {
        int d = mt * 16 + sel * 8 + r;
        out[d * 64 + ntp * 16 + laneM]       = acc0[r];
        out[d * 64 + (ntp + 1) * 16 + laneM] = acc1[r];
    }
}

// ---------------------------------------------------------------------------
// K6a: fold context into output weight:
//   Weff[b, o, h*64+d] = sum_e Wout[o, h*64+e] * ctx[b,h,d,e]   (f16 out)
// ---------------------------------------------------------------------------
__global__ void weff_kernel(const float* __restrict__ Wout,
                            const float* __restrict__ ctx,
                            half_t* __restrict__ Weff) {
    long idx = (long)blockIdx.x * blockDim.x + threadIdx.x;   // B*512*512
    int hd = (int)(idx & 511);
    long t1 = idx >> 9;
    int o = (int)(t1 & 511);
    int b = (int)(t1 >> 9);
    int h = hd >> 6, d = hd & 63;
    const float* w = Wout + (long)o * DIMC + h * DHEAD;
    const float* c = ctx + (((long)(b * HEADS + h) * 64) + d) * 64;
    float acc = 0.f;
    #pragma unroll
    for (int e = 0; e < 64; ++e) acc += w[e] * c[e];
    Weff[idx] = (half_t)acc;
}

// ---------------------------------------------------------------------------
// launcher
// ---------------------------------------------------------------------------
extern "C" void kernel_launch(void* const* d_in, const int* in_sizes, int n_in,
                              void* d_out, int out_size, void* d_ws, size_t ws_size,
                              hipStream_t stream) {
    const float* q     = (const float*)d_in[0];   // [B,512,T]
    const float* Wq    = (const float*)d_in[1];   // [512,512]
    const float* Wkv   = (const float*)d_in[2];   // [1024,512]
    const float* Wout  = (const float*)d_in[3];   // [512,512]
    const float* b_out = (const float*)d_in[4];   // [512]
    float* out = (float*)d_out;                   // [B,512,T]

    // workspace carve-up (bytes; all 256B-aligned). Total ~275 MB.
    char* ws = (char*)d_ws;
    constexpr long Q16_B  = (long)BATCH * DIMC * TDIM * 2;          //  67 MB
    constexpr long WC_B   = (long)PROWS * DIMC * 2;                 // 1.5 MB
    constexpr long P_B    = (long)BATCH * PROWS * TDIM * 2;         // 201 MB
    constexpr long CTX_B  = (long)BATCH * HEADS * 64 * 64 * 4;      //   1 MB
    half_t* q16  = (half_t*)(ws);
    half_t* Wc   = (half_t*)(ws + Q16_B);
    half_t* P    = (half_t*)(ws + Q16_B + WC_B);
    float*  ctx  = (float*) (ws + Q16_B + WC_B + P_B);
    half_t* Weff = (half_t*)(ws + Q16_B + WC_B + P_B + CTX_B);

    // K0: f32 -> f16 (q, and [Wq;Wkv] concatenated into Wc)
    {
        long nq = (long)BATCH * DIMC * TDIM;                        // 33,554,432
        cvt_f32_f16_kernel<<<(int)(nq / 2048), 256, 0, stream>>>(q, q16, nq);
        cvt_f32_f16_kernel<<<128, 256, 0, stream>>>(Wq,  Wc,              512L * 512);
        cvt_f32_f16_kernel<<<256, 256, 0, stream>>>(Wkv, Wc + 512L * 512, 1024L * 512);
    }

    // K1: projections  P[b] (1536 x T) = Wc (1536x512) * q16[b] (512 x T)
    gemm_f16_kernel<half_t, false><<<dim3(64, 12, BATCH), 256, 0, stream>>>(
        Wc, q16, P, nullptr, PROWS, (int)TDIM, DIMC,
        0L, (long)DIMC * TDIM, (long)PROWS * TDIM);

    // K2: softmax over d (in place, *SCALE) on qp rows
    qh_softmax_kernel<<<(int)((long)BATCH * HEADS * TDIM / 256), 256, 0, stream>>>(P);

    // K3: softmax over T (in place) on k rows
    kh_softmax_kernel<<<BATCH * 512, 256, 0, stream>>>(P);

    // K5: ctx[b,h] = kh * vh^T
    context_kernel<<<BATCH * HEADS, 256, 0, stream>>>(P, ctx);

    // K6a: Weff[b] = fold(Wout, ctx[b])
    weff_kernel<<<(int)((long)BATCH * DIMC * DIMC / 256), 256, 0, stream>>>(Wout, ctx, Weff);

    // K6b: y[b] = Weff[b] (512x512) * qh[b] (512 x T) + b_out
    gemm_f16_kernel<float, true><<<dim3(64, 4, BATCH), 256, 0, stream>>>(
        Weff, P, out, b_out, DIMC, (int)TDIM, DIMC,
        (long)DIMC * DIMC, (long)PROWS * TDIM, (long)DIMC * TDIM);
}
